// SS2D_with_SSD_4604204941450
// MI455X (gfx1250) — matrix operand order repaired and checked by reading, not compiled, not run on hardware
//
#include <hip/hip_runtime.h>
#include <hip/hip_bf16.h>

typedef __attribute__((ext_vector_type(16))) _Float16 v16h;
typedef __attribute__((ext_vector_type(8)))  _Float16 v8h;
typedef __attribute__((ext_vector_type(8)))  float    v8f;

#define WMMA16(a,b,c) __builtin_amdgcn_wmma_f32_16x16x32_f16(false,(a),false,(b),(short)0,(c),false,false)

constexpr int BSZ    = 4;
constexpr int HIMG   = 64;
constexpr int WIMG   = 64;
constexpr int LSEQ   = HIMG * WIMG;      // 4096
constexpr int DMODEL = 192;
constexpr int DSSM   = 384;
constexpr int DSTATE = 128;
constexpr int NH     = 6;                // heads per direction
constexpr int KDIR   = 4;
constexpr int CONVCH = DSSM + 2 * DSTATE + NH;          // 646
constexpr int DINPROJ= 2 * DSSM + 2 * DSTATE + NH;      // 1030
constexpr int CHUNK  = 256;
constexpr int NC     = LSEQ / CHUNK;     // 16
constexpr int PDIM   = 64;               // head dim

// ---------------------------------------------------------------- helpers ----
__device__ __forceinline__ float softplusf(float x){
  return x > 20.f ? x : log1pf(__expf(x));
}
__device__ __forceinline__ float siluf(float x){
  return x / (1.f + __expf(-x));
}
// direction re-index: identical for gather (build seq) and scatter (merge y)
__device__ __forceinline__ int src_index(int k, int l){
  int ll = (k & 2) ? (LSEQ - 1 - l) : l;
  return (k & 1) ? ((ll & (WIMG - 1)) * WIMG + (ll >> 6)) : ll;
}

// 16 contiguous halfs -> v16h (two 16B vector loads)
__device__ __forceinline__ v16h ld16(const _Float16* p){
  v8h lo = *(const v8h*)p;
  v8h hi = *(const v8h*)(p + 8);
  return __builtin_shufflevector(lo, hi, 0,1,2,3,4,5,6,7,8,9,10,11,12,13,14,15);
}

// A fragment 16x32 f16 from row-major [ld halfs] (rows=M, cols=K).
// lanes 0-15: K 0..15 ; lanes 16-31: K 16..31  -> contiguous per lane.
__device__ __forceinline__ v16h frag_a(const _Float16* p0, int row0, int col0, int ld){
  int l = threadIdx.x & 31;
  const _Float16* p = p0 + (size_t)(row0 + (l & 15)) * ld + col0 + ((l >> 4) << 4);
  v8h lo = *(const v8h*)p;
  v8h hi = *(const v8h*)(p + 8);
  return __builtin_shufflevector(lo, hi, 0,1,2,3,4,5,6,7,8,9,10,11,12,13,14,15);
}
// B fragment 32x16 where memory holds the TRANSPOSE: mem[col][k] row-major [ld].
// lanes 0-15: K 0..15 ; lanes 16-31: K 16..31  -> contiguous per lane.
__device__ __forceinline__ v16h frag_bT(const _Float16* p0, int col0, int k0, int ld){
  int l = threadIdx.x & 31;
  return ld16(p0 + (size_t)(col0 + (l & 15)) * ld + k0 + ((l >> 4) << 4));
}

// --------------------------- CDNA5 async global->LDS copy (ASYNCcnt) ----
__device__ __forceinline__ void async_ld_b128(unsigned ldsoff, const void* g){
  asm volatile("global_load_async_to_lds_b128 %0, %1, off"
               :: "v"(ldsoff), "v"(g) : "memory");
}
__device__ __forceinline__ void wait_async0(){
  asm volatile("s_wait_asynccnt 0x0" ::: "memory");
}

// ---------------------------------------------------------------- utility ----
__global__ void cast_f32_f16(const float* __restrict__ in, _Float16* __restrict__ out, size_t n){
  size_t i = (size_t)blockIdx.x * 256 + threadIdx.x;
  if (i < n) out[i] = (_Float16)in[i];
}
// in[K][N] f32 -> out[N][K] f16
__global__ void cast_transpose_f16(const float* __restrict__ in, _Float16* __restrict__ out,
                                   int Kd, int N){
  size_t i = (size_t)blockIdx.x * 256 + threadIdx.x;
  if (i >= (size_t)Kd * N) return;
  int n = (int)(i / Kd), k = (int)(i % Kd);
  out[i] = (_Float16)in[(size_t)k * N + n];
}
__global__ void zero_f32(float* __restrict__ p, size_t n){
  size_t i = (size_t)blockIdx.x * 256 + threadIdx.x;
  if (i < n) p[i] = 0.f;
}

// ------------------------------------------------- generic WMMA f16 GEMM ----
// C[M,N] f32 = A[M,K] f16 * Bt[N,K] f16 (B pre-transposed).
// grid = (ceil(N/128), M/32), block 128 (4 waves, each 32 rows x 32 cols).
__global__ void __launch_bounds__(128)
wmma_gemm(const _Float16* __restrict__ A, const _Float16* __restrict__ Bt,
          float* __restrict__ C, int M, int N, int Kd){
  int w = threadIdx.x >> 5, lane = threadIdx.x & 31;
  int l15 = lane & 15, hi8 = (lane >> 4) << 3, hi16 = (lane >> 4) << 4;
  int m0 = blockIdx.y * 32;
  int nb = blockIdx.x * 128 + w * 32;
  v8f acc[2][2];
#pragma unroll
  for (int im = 0; im < 2; ++im)
#pragma unroll
    for (int jn = 0; jn < 2; ++jn) acc[im][jn] = (v8f){};
  for (int k0 = 0; k0 < Kd; k0 += 32){
    if (k0 + 32 < Kd){
      __builtin_prefetch(A + (size_t)(m0 + (lane & 15)) * Kd + k0 + 32, 0, 1);
      __builtin_prefetch(Bt + (size_t)(nb + (lane & 15)) * Kd + k0 + 32, 0, 1);
    }
    v16h a0 = frag_a(A, m0,      k0, Kd);
    v16h a1 = frag_a(A, m0 + 16, k0, Kd);
#pragma unroll
    for (int jn = 0; jn < 2; ++jn){
      int ncl = nb + jn * 16 + l15; if (ncl >= N) ncl = N - 1;   // clamp OOB cols
      v16h bb = ld16(Bt + (size_t)ncl * Kd + k0 + hi16);
      acc[0][jn] = WMMA16(a0, bb, acc[0][jn]);
      acc[1][jn] = WMMA16(a1, bb, acc[1][jn]);
    }
  }
#pragma unroll
  for (int im = 0; im < 2; ++im)
#pragma unroll
    for (int jn = 0; jn < 2; ++jn)
#pragma unroll
      for (int r = 0; r < 8; ++r){
        int m = m0 + im * 16 + r + hi8, n = nb + jn * 16 + l15;
        if (n < N) C[(size_t)m * N + n] = acc[im][jn][r];
      }
}

// ------------------------------------------- depthwise 3x3 conv + SiLU ----
__global__ void k2_conv(const float* __restrict__ zx, const float* __restrict__ cw,
                        const float* __restrict__ cb, float* __restrict__ xBCc){
  size_t i = (size_t)blockIdx.x * 256 + threadIdx.x;
  if (i >= (size_t)BSZ * LSEQ * CONVCH) return;
  int c  = (int)(i % CONVCH);
  size_t bl = i / CONVCH;
  int l = (int)(bl % LSEQ);
  int b = (int)(bl / LSEQ);
  int y = l >> 6, x = l & 63;
  float s = cb[c];
#pragma unroll
  for (int dy = 0; dy < 3; ++dy){
    int yy = y + dy - 1; if ((unsigned)yy >= (unsigned)HIMG) continue;
#pragma unroll
    for (int dx = 0; dx < 3; ++dx){
      int xx = x + dx - 1; if ((unsigned)xx >= (unsigned)WIMG) continue;
      s += zx[((size_t)b * LSEQ + yy * WIMG + xx) * DINPROJ + DSSM + c] * cw[c * 9 + dy * 3 + dx];
    }
  }
  xBCc[i] = siluf(s);
}

// -------------------------- per (b,dir,head,chunk) cumsum of dA = dt*A ----
__global__ void __launch_bounds__(256)
k3a_dAc(const float* __restrict__ xBCc, const float* __restrict__ dt_bias,
        const float* __restrict__ A_logs, float* __restrict__ dAc){
  __shared__ float sh[CHUNK];
  int z = blockIdx.x, h = blockIdx.y, bk = blockIdx.z;
  int b = bk >> 2, k = bk & 3, hh = k * NH + h;
  int c = threadIdx.x, l = z * CHUNK + c;
  int srci = src_index(k, l);
  float raw = xBCc[((size_t)b * LSEQ + srci) * CONVCH + DSSM + 2 * DSTATE + h];
  float dtv = softplusf(raw + dt_bias[hh]);
  sh[c] = -__expf(A_logs[hh]) * dtv;
  __syncthreads();
  for (int off = 1; off < CHUNK; off <<= 1){
    float v = (c >= off) ? sh[c - off] : 0.f;
    __syncthreads();
    sh[c] += v;
    __syncthreads();
  }
  dAc[((size_t)(bk * NH + h)) * LSEQ + l] = sh[c];
}

// ---------------- build direction sequences in f16: x~=x*dt, B, C ----
__global__ void k3b_seq(const float* __restrict__ xBCc, const float* __restrict__ dt_bias,
                        _Float16* __restrict__ Xtil, _Float16* __restrict__ Bseq,
                        _Float16* __restrict__ Cseq){
  size_t i = (size_t)blockIdx.x * 256 + threadIdx.x;
  size_t total = (size_t)BSZ * KDIR * LSEQ * (DSSM + 2 * DSTATE);
  if (i >= total) return;
  int c = (int)(i % (DSSM + 2 * DSTATE));
  size_t t = i / (DSSM + 2 * DSTATE);
  int l = (int)(t % LSEQ);
  int bk = (int)(t / LSEQ);
  int b = bk >> 2, k = bk & 3;
  int srci = src_index(k, l);
  size_t base = ((size_t)b * LSEQ + srci) * CONVCH;
  size_t row  = (size_t)(bk * LSEQ + l);
  if (c < DSSM){
    int h = c >> 6;
    float raw = xBCc[base + DSSM + 2 * DSTATE + h];
    float dtv = softplusf(raw + dt_bias[k * NH + h]);
    Xtil[row * DSSM + c] = (_Float16)(xBCc[base + c] * dtv);
  } else if (c < DSSM + DSTATE){
    Bseq[row * DSTATE + (c - DSSM)] = (_Float16)xBCc[base + c];
  } else {
    Cseq[row * DSTATE + (c - DSSM - DSTATE)] = (_Float16)xBCc[base + c];
  }
}

// ------------- chunk end-states: states[p,n] = sum_s x~dec[s,p] B[s,n] ----
__global__ void __launch_bounds__(256)
k4_states(const _Float16* __restrict__ Xtil, const _Float16* __restrict__ Bseq,
          const float* __restrict__ dAc, float* __restrict__ states){
  __shared__ __align__(16) _Float16 Axt [PDIM][32];     // x~dec^T (p,s)  4 KB
  __shared__ __align__(16) _Float16 BshT[DSTATE][32];   // B^T (n,s)      8 KB
  int z = blockIdx.x, h = blockIdx.y, bk = blockIdx.z;
  int tid = threadIdx.x, w = tid >> 5, lane = tid & 31;
  int l15 = lane & 15, hi8 = (lane >> 4) << 3;
  size_t crow  = (size_t)(bk * LSEQ + z * CHUNK);
  size_t dbase = ((size_t)(bk * NH + h)) * LSEQ + z * CHUNK;
  float dlast = dAc[dbase + CHUNK - 1];
  int pt = w & 3, ng = w >> 2;
  v8f acc[4];
#pragma unroll
  for (int j = 0; j < 4; ++j) acc[j] = (v8f){};
  for (int s0 = 0; s0 < CHUNK; s0 += 32){
    __syncthreads();
    {
      int s = (tid * 8) >> 6;                 // one s per thread-octet
      int p = (tid * 8) & 63;
      float dec = __expf(dlast - dAc[dbase + s0 + s]);
#pragma unroll
      for (int e = 0; e < 8; ++e)
        Axt[p + e][s] = (_Float16)((float)Xtil[(crow + s0 + s) * DSSM + h * PDIM + p + e] * dec);
    }
    {
      int s = (tid * 16) >> 7;                // contiguous 16-half global read
      int n = (tid * 16) & 127;
#pragma unroll
      for (int e = 0; e < 16; ++e)
        BshT[n + e][s] = Bseq[(crow + s0 + s) * DSTATE + n + e];
    }
    __syncthreads();
    v16h a = frag_a(&Axt[0][0], 16 * pt, 0, 32);
#pragma unroll
    for (int j = 0; j < 4; ++j){
      v16h bb = frag_bT(&BshT[0][0], 16 * (ng * 4 + j), 0, 32);
      acc[j] = WMMA16(a, bb, acc[j]);
    }
  }
  size_t sb = (((size_t)(bk * NH + h)) * NC + z) * (PDIM * DSTATE);
#pragma unroll
  for (int j = 0; j < 4; ++j)
#pragma unroll
    for (int r = 0; r < 8; ++r){
      int p = 16 * pt + r + hi8, n = 16 * (ng * 4 + j) + l15;
      states[sb + (size_t)p * DSTATE + n] = acc[j][r];
    }
}

// ----------- sequential inter-chunk scan (in-place: states -> prev) ----
__global__ void __launch_bounds__(256)
k5_scan(const float* __restrict__ dAc, float* __restrict__ states){
  int g = blockIdx.x, tid = threadIdx.x;          // g = bk*NH+h
  float carry[32];
#pragma unroll
  for (int i = 0; i < 32; ++i) carry[i] = 0.f;
  size_t gb = (size_t)g * NC * PDIM * DSTATE;
  size_t db = (size_t)g * LSEQ;
  for (int z = 0; z < NC; ++z){
    float cd = __expf(dAc[db + z * CHUNK + CHUNK - 1]);
    size_t zb = gb + (size_t)z * PDIM * DSTATE;
#pragma unroll
    for (int i = 0; i < 32; ++i){
      size_t idx = zb + tid + i * 256;
      float s = states[idx];
      states[idx] = carry[i];
      carry[i] = carry[i] * cd + s;
    }
  }
}

// -------------------- SSD core: intra (CB∘seg)·x~ + inter C·prev + D·x ----
__global__ void __launch_bounds__(256)
k6_ssd(const _Float16* __restrict__ Cseq, const _Float16* __restrict__ Bseq,
       const _Float16* __restrict__ Xtil, const float* __restrict__ dAc,
       const float* __restrict__ prev, const float* __restrict__ xBCc,
       const float* __restrict__ Ds, float* __restrict__ out_pre){
  __shared__ __align__(16) float    dAs[CHUNK];            //  1 KB
  __shared__ __align__(16) _Float16 prevSh[PDIM][DSTATE];  // 16 KB (p,n) as in memory
  __shared__ __align__(16) _Float16 CBseg[CHUNK][32];      // 16 KB (c, s-strip)
  __shared__ __align__(16) _Float16 XshT[PDIM][32];        //  4 KB (p,s) strip
  __shared__ __align__(16) _Float16 Bsl[2][32][DSTATE];    // 16 KB async-staged B strips
  int z = blockIdx.x, h = blockIdx.y, bk = blockIdx.z;
  int b = bk >> 2, k = bk & 3, hh = k * NH + h;
  int tid = threadIdx.x, w = tid >> 5, lane = tid & 31;
  int l15 = lane & 15, hi8 = (lane >> 4) << 3;
  size_t crow = (size_t)(bk * LSEQ + z * CHUNK);
  const _Float16* Cg = Cseq + crow * DSTATE;
  const _Float16* Bg = Bseq + crow * DSTATE;
  size_t dbase = ((size_t)(bk * NH + h)) * LSEQ + z * CHUNK;
  dAs[tid] = dAc[dbase + tid];
  size_t pb = (((size_t)(bk * NH + h)) * NC + z) * (PDIM * DSTATE);
  for (int i = tid; i < PDIM * DSTATE; i += 256)
    prevSh[i >> 7][i & 127] = (_Float16)prev[pb + i];

  // LDS byte offsets of the two async-staging buffers (flat LDS addr[31:0])
  unsigned bsl0 = (unsigned)(size_t)(void*)&Bsl[0][0][0] + tid * 32u;
  unsigned bsl1 = (unsigned)(size_t)(void*)&Bsl[1][0][0] + tid * 32u;
  // issue async copy of strip 0 (32x128 f16 = 8KB; 2 x b128 per lane)
  {
    const char* g = (const char*)Bg + tid * 32u;
    async_ld_b128(bsl0,      g);
    async_ld_b128(bsl0 + 16, g + 16);
  }
  __syncthreads();

  // preload this wave's C A-fragments (re-used by inter and every CB strip)
  v16h afr[2][4];
#pragma unroll
  for (int ct = 0; ct < 2; ++ct)
#pragma unroll
    for (int kk = 0; kk < 4; ++kk)
      afr[ct][kk] = frag_a(Cg, 16 * (w + 8 * ct), 32 * kk, DSTATE);

  v8f acc[2][4];
#pragma unroll
  for (int ct = 0; ct < 2; ++ct)
#pragma unroll
    for (int j = 0; j < 4; ++j) acc[ct][j] = (v8f){};

  // inter: y[c,p] = sum_n C[c,n] * prev[p,n]   (prevSh is [p][n] == B^T form)
#pragma unroll
  for (int ct = 0; ct < 2; ++ct)
    for (int kk = 0; kk < 4; ++kk)
#pragma unroll
      for (int j = 0; j < 4; ++j){
        v16h bb = frag_bT(&prevSh[0][0], 16 * j, 32 * kk, DSTATE);
        acc[ct][j] = WMMA16(afr[ct][kk], bb, acc[ct][j]);
      }
  // scale inter by exp(dAc[c])
#pragma unroll
  for (int ct = 0; ct < 2; ++ct){
    int c0 = 16 * (w + 8 * ct);
#pragma unroll
    for (int j = 0; j < 4; ++j)
#pragma unroll
      for (int r = 0; r < 8; ++r)
        acc[ct][j][r] *= __expf(dAs[c0 + r + hi8]);
  }

  // intra over 8 s-strips of 32 (double-buffered async B staging)
  for (int ss = 0; ss < 8; ++ss){
    wait_async0();          // own portion of strip ss landed in LDS (in-order)
    __syncthreads();        // all waves' portions visible; prior LDS reads done
    if (ss + 1 < 8){        // stream next strip while computing this one
      unsigned dst = ((ss + 1) & 1) ? bsl1 : bsl0;
      const char* g = (const char*)(Bg + (size_t)(ss + 1) * 32 * DSTATE) + tid * 32u;
      async_ld_b128(dst,      g);
      async_ld_b128(dst + 16, g + 16);
    }
    const _Float16* Bs = &Bsl[ss & 1][0][0];
    // stage x~ strip transposed: XshT[p][s]
    {
      int p = (tid * 8) & 63, s = (tid * 8) >> 6;
#pragma unroll
      for (int e = 0; e < 8; ++e)
        XshT[p + e][s] = Xtil[(crow + ss * 32 + s) * DSSM + h * PDIM + p + e];
    }
    // CB tiles (c,s) with segment mask/exp, written to LDS f16
#pragma unroll
    for (int ct2 = 0; ct2 < 2; ++ct2){
      int c0 = 16 * (w + 8 * ct2);
      v8f cb[2] = { (v8f){}, (v8f){} };
      for (int kk = 0; kk < 4; ++kk)
#pragma unroll
        for (int st = 0; st < 2; ++st){
          v16h bb = frag_bT(Bs, 16 * st, 32 * kk, DSTATE);
          cb[st] = WMMA16(afr[ct2][kk], bb, cb[st]);
        }
#pragma unroll
      for (int st = 0; st < 2; ++st)
#pragma unroll
        for (int r = 0; r < 8; ++r){
          int c = c0 + r + hi8, s = ss * 32 + 16 * st + l15;
          float v = (s <= c) ? cb[st][r] * __expf(dAs[c] - dAs[s]) : 0.f;
          CBseg[c][s - ss * 32] = (_Float16)v;
        }
    }
    __syncthreads();
#pragma unroll
    for (int ct = 0; ct < 2; ++ct){
      v16h a = frag_a(&CBseg[0][0], 16 * (w + 8 * ct), 0, 32);
#pragma unroll
      for (int j = 0; j < 4; ++j){
        v16h bb = frag_bT(&XshT[0][0], 16 * j, 0, 32);
        acc[ct][j] = WMMA16(a, bb, acc[ct][j]);
      }
    }
  }

  // epilogue: + D*x, scatter-add into direction-merged buffer
  float dsv = Ds[hh];
#pragma unroll
  for (int ct = 0; ct < 2; ++ct){
    int c0 = 16 * (w + 8 * ct);
#pragma unroll
    for (int j = 0; j < 4; ++j)
#pragma unroll
      for (int r = 0; r < 8; ++r){
        int c = c0 + r + hi8, p = 16 * j + l15;
        int lg = z * CHUNK + c;
        int srci = src_index(k, lg);
        float xv = xBCc[((size_t)b * LSEQ + srci) * CONVCH + h * PDIM + p];
        atomicAdd(&out_pre[((size_t)b * LSEQ + srci) * DSSM + h * PDIM + p],
                  acc[ct][j][r] + dsv * xv);
      }
  }
}

// --------------------------- gated SiLU + RMS-norm, emit f16 rows ----
__global__ void __launch_bounds__(128)
k7a_norm(const float* __restrict__ out_pre, const float* __restrict__ zx,
         const float* __restrict__ norm_w, _Float16* __restrict__ gbuf){
  __shared__ float red[128];
  int row = blockIdx.x, tid = threadIdx.x;
  float g[3], ss = 0.f;
#pragma unroll
  for (int i = 0; i < 3; ++i){
    int c = tid + i * 128;
    float o  = out_pre[(size_t)row * DSSM + c];
    float zv = zx[(size_t)row * DINPROJ + c];
    float gg = o * siluf(zv);
    g[i] = gg; ss += gg * gg;
  }
  red[tid] = ss; __syncthreads();
  for (int o = 64; o > 0; o >>= 1){
    if (tid < o) red[tid] += red[tid + o];
    __syncthreads();
  }
  float r = rsqrtf(red[0] / DSSM + 1e-5f);
#pragma unroll
  for (int i = 0; i < 3; ++i){
    int c = tid + i * 128;
    gbuf[(size_t)row * DSSM + c] = (_Float16)(g[i] * r * norm_w[c]);
  }
}

// ------------------------------------------------------------- launch ----
extern "C" void kernel_launch(void* const* d_in, const int* in_sizes, int n_in,
                              void* d_out, int out_size, void* d_ws, size_t ws_size,
                              hipStream_t stream){
  const float* u       = (const float*)d_in[0];
  const float* W_in    = (const float*)d_in[1];
  const float* conv_w  = (const float*)d_in[2];
  const float* conv_b  = (const float*)d_in[3];
  const float* dt_bias = (const float*)d_in[4];
  const float* A_logs  = (const float*)d_in[5];
  const float* Ds      = (const float*)d_in[6];
  const float* norm_w  = (const float*)d_in[7];
  const float* W_out   = (const float*)d_in[8];

  char* ws = (char*)d_ws;
  auto carve = [&](size_t bytes) -> void* {
    void* p = (void*)ws;
    ws += (bytes + 255) & ~(size_t)255;
    return p;
  };
  const size_t M = (size_t)BSZ * LSEQ;                      // 16384 rows
  float*    zx     = (float*)   carve(M * DINPROJ * 4);
  float*    xBCc   = (float*)   carve(M * CONVCH * 4);
  _Float16* uh     = (_Float16*)carve(M * DMODEL * 2);
  _Float16* Winh   = (_Float16*)carve((size_t)DMODEL * DINPROJ * 2);   // [N][K]
  _Float16* Wouth  = (_Float16*)carve((size_t)DSSM * DMODEL * 2);      // [N][K]
  _Float16* Xtil   = (_Float16*)carve((size_t)BSZ * KDIR * LSEQ * DSSM * 2);
  _Float16* Bseq   = (_Float16*)carve((size_t)BSZ * KDIR * LSEQ * DSTATE * 2);
  _Float16* Cseq   = (_Float16*)carve((size_t)BSZ * KDIR * LSEQ * DSTATE * 2);
  float*    dAc    = (float*)   carve((size_t)BSZ * KDIR * NH * LSEQ * 4);
  float*    states = (float*)   carve((size_t)BSZ * KDIR * NH * NC * PDIM * DSTATE * 4);
  float*    outp   = (float*)   carve(M * DSSM * 4);
  _Float16* gbuf   = (_Float16*)carve(M * DSSM * 2);

  // casts for GEMM operands (weights transposed to [N][K])
  size_t nu = M * DMODEL;
  cast_f32_f16<<<(unsigned)((nu + 255) / 256), 256, 0, stream>>>(u, uh, nu);
  size_t nwi = (size_t)DMODEL * DINPROJ;
  cast_transpose_f16<<<(unsigned)((nwi + 255) / 256), 256, 0, stream>>>(W_in, Winh, DMODEL, DINPROJ);
  size_t nwo = (size_t)DSSM * DMODEL;
  cast_transpose_f16<<<(unsigned)((nwo + 255) / 256), 256, 0, stream>>>(W_out, Wouth, DSSM, DMODEL);

  // in-projection GEMM (16384 x 1030 x 192)
  wmma_gemm<<<dim3((DINPROJ + 127) / 128, (unsigned)(M / 32)), 128, 0, stream>>>(
      uh, Winh, zx, (int)M, DINPROJ, DMODEL);

  // depthwise conv + SiLU
  size_t nc2 = M * CONVCH;
  k2_conv<<<(unsigned)((nc2 + 255) / 256), 256, 0, stream>>>(zx, conv_w, conv_b, xBCc);

  // dA cumsum per chunk; direction sequences
  k3a_dAc<<<dim3(NC, NH, BSZ * KDIR), 256, 0, stream>>>(xBCc, dt_bias, A_logs, dAc);
  size_t nseq = (size_t)BSZ * KDIR * LSEQ * (DSSM + 2 * DSTATE);
  k3b_seq<<<(unsigned)((nseq + 255) / 256), 256, 0, stream>>>(xBCc, dt_bias, Xtil, Bseq, Cseq);

  // chunk states, inter-chunk scan
  k4_states<<<dim3(NC, NH, BSZ * KDIR), 256, 0, stream>>>(Xtil, Bseq, dAc, states);
  k5_scan<<<BSZ * KDIR * NH, 256, 0, stream>>>(dAc, states);

  // merged output accumulator must be zeroed every call (atomics)
  size_t nop = M * DSSM;
  zero_f32<<<(unsigned)((nop + 255) / 256), 256, 0, stream>>>(outp, nop);

  // SSD core
  k6_ssd<<<dim3(NC, NH, BSZ * KDIR), 256, 0, stream>>>(
      Cseq, Bseq, Xtil, dAc, states, xBCc, Ds, outp);

  // gate + RMS-norm, out-projection GEMM (16384 x 192 x 384)
  k7a_norm<<<(unsigned)M, 128, 0, stream>>>(outp, zx, norm_w, gbuf);
  wmma_gemm<<<dim3((DMODEL + 127) / 128, (unsigned)(M / 32)), 128, 0, stream>>>(
      gbuf, Wouth, (float*)d_out, (int)M, DMODEL, DSSM);
}
